// GPTModel_49289044689000
// MI455X (gfx1250) — compile-verified, matrix operand order checked by reading, not run on hardware
//
#include <hip/hip_runtime.h>
#include <hip/hip_bf16.h>

typedef __attribute__((ext_vector_type(16))) __bf16 v16bf;
typedef __attribute__((ext_vector_type(8)))  float  v8f;

#define B_   2
#define T_   1024
#define D_   1024
#define H_   16
#define F_   4096
#define L_   8
#define V_   32000
#define HD_  64
#define M_   (B_ * T_)   // 2048 tokens

static_assert(M_ % 128 == 0 && D_ % 128 == 0 && F_ % 128 == 0 && V_ % 128 == 0, "tiles");

union Frag { uint4 u[2]; v16bf v; };   // 16 bf16 = one lane's WMMA A/B operand
union Acc  { v8f v; float f[8]; };     // 16x16 f32 C/D tile slice per lane

static __device__ __forceinline__ v8f wmma_bf16(v16bf a, v16bf b, v8f c) {
  // (neg_a, A, neg_b, B, c_mod, C, reuse_a, reuse_b)
  return __builtin_amdgcn_wmma_f32_16x16x32_bf16(false, a, false, b, (short)0, c,
                                                 false, false);
}

// ---------------------------------------------------------------------------
// x[b,t,:] = tok_emb[idx[b,t],:] + pos_emb[t,:]
// ---------------------------------------------------------------------------
__global__ __launch_bounds__(256) void embed_kernel(const int* __restrict__ idx,
                                                    const float* __restrict__ tok,
                                                    const float* __restrict__ pos,
                                                    float* __restrict__ x) {
  const int row = blockIdx.x;          // 0..M_-1
  const int t   = row % T_;
  const int tk  = idx[row];
  const float* te = tok + (size_t)tk * D_;
  const float* pe = pos + (size_t)t * D_;
  float* xr = x + (size_t)row * D_;
  for (int c = threadIdx.x; c < D_; c += 256) xr[c] = te[c] + pe[c];
}

// ---------------------------------------------------------------------------
// LayerNorm over D, f32 in, bf16 out (feeds WMMA GEMMs).
// ---------------------------------------------------------------------------
__global__ __launch_bounds__(256) void layernorm_kernel(const float* __restrict__ x,
                                                        const float* __restrict__ g,
                                                        const float* __restrict__ b,
                                                        __bf16* __restrict__ out) {
  __shared__ float red[256];
  const int row = blockIdx.x;
  const int tid = threadIdx.x;
  const float* xr = x + (size_t)row * D_;
  float vals[4];
  float s = 0.f, ss = 0.f;
  for (int j = 0; j < 4; ++j) {
    float v = xr[tid + j * 256];
    vals[j] = v; s += v; ss += v * v;
  }
  red[tid] = s; __syncthreads();
  for (int o = 128; o > 0; o >>= 1) {
    if (tid < o) red[tid] += red[tid + o];
    __syncthreads();
  }
  const float mean = red[0] * (1.0f / D_);
  __syncthreads();
  red[tid] = ss; __syncthreads();
  for (int o = 128; o > 0; o >>= 1) {
    if (tid < o) red[tid] += red[tid + o];
    __syncthreads();
  }
  const float var = red[0] * (1.0f / D_) - mean * mean;
  const float inv = rsqrtf(var + 1e-5f);
  __bf16* orow = out + (size_t)row * D_;
  for (int j = 0; j < 4; ++j) {
    int c = tid + j * 256;
    orow[c] = (__bf16)((vals[j] - mean) * inv * g[c] + b[c]);
  }
}

// ---------------------------------------------------------------------------
// W [KxN] f32 row-major  ->  W^T [NxK] bf16 (so GEMM B-tiles stage contiguous).
// ---------------------------------------------------------------------------
__global__ __launch_bounds__(256) void transpose_cvt_kernel(const float* __restrict__ in,
                                                            __bf16* __restrict__ out,
                                                            int K, int N) {
  __shared__ float ts[32][33];
  const int kb = blockIdx.y * 32, nb = blockIdx.x * 32;
  const int tx = threadIdx.x, ty = threadIdx.y;   // (32,8)
  for (int j = 0; j < 4; ++j)
    ts[ty + j * 8][tx] = in[(size_t)(kb + ty + j * 8) * N + nb + tx];
  __syncthreads();
  for (int j = 0; j < 4; ++j)
    out[(size_t)(nb + ty + j * 8) * K + kb + tx] = (__bf16)ts[tx][ty + j * 8];
}

// ---------------------------------------------------------------------------
// GEMM: C[M,N] = A[M,K](bf16) * Bt[N,K](bf16)^T + bias, epilogue options.
// Block: 128x128 tile, 256 threads = 8 waves (4x2), wave tile 32x64 (2x4 WMMA).
// Double-buffered LDS (2 x 14 KiB per matrix): global loads for tile i+1 are
// issued before the WMMAs of tile i, so load latency overlaps matrix math.
// LDS row stride 56 bf16 (112B): 16B aligned, conflict-free fragment reads.
// ---------------------------------------------------------------------------
template <bool RES, bool GELU, bool OUTF, bool OUTB>
__global__ __launch_bounds__(256) void gemm_kernel(const __bf16* __restrict__ A,
                                                   const __bf16* __restrict__ Bt,
                                                   const float* __restrict__ bias,
                                                   const float* __restrict__ res,
                                                   float* __restrict__ outF,
                                                   __bf16* __restrict__ outB,
                                                   int M, int N, int K) {
  constexpr int LDT  = 56;
  constexpr int TSZ  = 128 * LDT;
  __shared__ __align__(16) __bf16 As[2 * TSZ];
  __shared__ __align__(16) __bf16 Bs[2 * TSZ];
  (void)M;
  const int tid  = threadIdx.x;
  const int wave = tid >> 5, lane = tid & 31;
  const int half = lane >> 4, lp = lane & 15;
  const int wm = wave >> 1, wn = wave & 1;
  const int m0 = blockIdx.y * 128, n0 = blockIdx.x * 128;

  // Per-thread staging coordinates: 512 16B chunks per matrix, 2 per thread.
  const int r0 = tid >> 2,            cc0 = (tid & 3) * 8;
  const int r1 = (tid + 256) >> 2,    cc1 = ((tid + 256) & 3) * 8;
  const __bf16* Arow0 = A  + (size_t)(m0 + r0) * K + cc0;
  const __bf16* Arow1 = A  + (size_t)(m0 + r1) * K + cc1;
  const __bf16* Brow0 = Bt + (size_t)(n0 + r0) * K + cc0;
  const __bf16* Brow1 = Bt + (size_t)(n0 + r1) * K + cc1;

  Acc acc[2][4];
  for (int mt = 0; mt < 2; ++mt)
    for (int nt = 0; nt < 4; ++nt)
      for (int i = 0; i < 8; ++i) acc[mt][nt].f[i] = 0.f;

  // Prologue: stage K-tile 0 into buffer 0.
  {
    uint4 a0 = *(const uint4*)(Arow0);
    uint4 a1 = *(const uint4*)(Arow1);
    uint4 b0 = *(const uint4*)(Brow0);
    uint4 b1 = *(const uint4*)(Brow1);
    *(uint4*)(&As[r0 * LDT + cc0]) = a0;
    *(uint4*)(&As[r1 * LDT + cc1]) = a1;
    *(uint4*)(&Bs[r0 * LDT + cc0]) = b0;
    *(uint4*)(&Bs[r1 * LDT + cc1]) = b1;
  }

  for (int k0 = 0; k0 < K; k0 += 32) {
    const int cur = (k0 >> 5) & 1;
    const int nxt = cur ^ 1;
    __syncthreads();   // buffer `cur` fully staged; prior reads of `nxt` done

    // Issue global loads for the next K tile (in flight during the WMMAs).
    uint4 a0, a1, b0, b1;
    const bool more = (k0 + 32) < K;
    if (more) {
      a0 = *(const uint4*)(Arow0 + k0 + 32);
      a1 = *(const uint4*)(Arow1 + k0 + 32);
      b0 = *(const uint4*)(Brow0 + k0 + 32);
      b1 = *(const uint4*)(Brow1 + k0 + 32);
    }
    if (k0 + 64 < K) {  // prefetch two tiles ahead (global_prefetch_b8)
      int r = tid & 127;
      if (tid < 128) __builtin_prefetch((const void*)(A  + (size_t)(m0 + r) * K + k0 + 64), 0, 1);
      else           __builtin_prefetch((const void*)(Bt + (size_t)(n0 + r) * K + k0 + 64), 0, 1);
    }

    // Fragment loads per ISA 16-bit A/B layout:
    //   lane: row = lp, chunk0 = [half*8 .. +7], chunk1 = [16+half*8 .. +7]
    const __bf16* asb = As + cur * TSZ;
    const __bf16* bsb = Bs + cur * TSZ;
    Frag a[2], b[4];
    for (int mt = 0; mt < 2; ++mt) {
      const __bf16* p = asb + (wm * 32 + mt * 16 + lp) * LDT;
      a[mt].u[0] = *(const uint4*)(p + half * 8);
      a[mt].u[1] = *(const uint4*)(p + 16 + half * 8);
    }
    for (int nt = 0; nt < 4; ++nt) {
      const __bf16* p = bsb + (wn * 64 + nt * 16 + lp) * LDT;
      b[nt].u[0] = *(const uint4*)(p + half * 8);
      b[nt].u[1] = *(const uint4*)(p + 16 + half * 8);
    }
    for (int mt = 0; mt < 2; ++mt)
      for (int nt = 0; nt < 4; ++nt)
        acc[mt][nt].v = wmma_bf16(a[mt].v, b[nt].v, acc[mt][nt].v);

    // Store the staged registers into the other buffer for the next round.
    if (more) {
      __bf16* asn = As + nxt * TSZ;
      __bf16* bsn = Bs + nxt * TSZ;
      *(uint4*)(&asn[r0 * LDT + cc0]) = a0;
      *(uint4*)(&asn[r1 * LDT + cc1]) = a1;
      *(uint4*)(&bsn[r0 * LDT + cc0]) = b0;
      *(uint4*)(&bsn[r1 * LDT + cc1]) = b1;
    }
  }

  // Epilogue: C layout -> lane holds col lp, rows i + 8*half.
  for (int mt = 0; mt < 2; ++mt)
    for (int nt = 0; nt < 4; ++nt) {
      const int col = n0 + wn * 64 + nt * 16 + lp;
      const float bv = bias[col];
      for (int i = 0; i < 8; ++i) {
        const int row = m0 + wm * 32 + mt * 16 + i + 8 * half;
        float v = acc[mt][nt].f[i] + bv;
        if (RES)  v += res[(size_t)row * N + col];
        if (GELU) v = 0.5f * v * (1.0f + erff(v * 0.70710678118654752f));
        if (OUTF) outF[(size_t)row * N + col] = v;
        if (OUTB) outB[(size_t)row * N + col] = (__bf16)v;
      }
    }
}

// ---------------------------------------------------------------------------
// Flash attention: grid (T/128 q-blocks, H, B), 256 threads = 8 waves.
// Wave w owns 16 q-rows; streams 128-key blocks with online softmax.
// K B-fragments are software-pipelined 2 deep so global load latency overlaps
// the QK^T WMMAs.
// ---------------------------------------------------------------------------
__global__ __launch_bounds__(256) void attention_kernel(const __bf16* __restrict__ q,
                                                        const __bf16* __restrict__ k,
                                                        const __bf16* __restrict__ v,
                                                        __bf16* __restrict__ o) {
  __shared__ __align__(16) __bf16 Vs[64][136];       // V^T tile  [hd][key]
  __shared__ __align__(16) __bf16 Ps[8][16][136];    // per-wave P strip
  const int qb = blockIdx.x, hh = blockIdx.y, bb = blockIdx.z;
  const int tid = threadIdx.x;
  const int wave = tid >> 5, lane = tid & 31;
  const int half = lane >> 4, lp = lane & 15;
  const size_t base = (size_t)bb * T_ * D_ + hh * HD_;
  const float scale = 0.125f;                        // 1/sqrt(64)
  const float NEG = -3.0e38f;

  Frag qf[2];
  {
    const __bf16* qp = q + base + (size_t)(qb * 128 + wave * 16 + lp) * D_;
    for (int kk = 0; kk < 2; ++kk) {
      qf[kk].u[0] = *(const uint4*)(qp + kk * 32 + half * 8);
      qf[kk].u[1] = *(const uint4*)(qp + kk * 32 + 16 + half * 8);
    }
  }

  Acc oacc[4];
  float mrow[8], lrow[8];
  for (int nt = 0; nt < 4; ++nt)
    for (int i = 0; i < 8; ++i) oacc[nt].f[i] = 0.f;
  for (int i = 0; i < 8; ++i) { mrow[i] = NEG; lrow[i] = 0.f; }

  for (int kb = 0; kb <= qb; ++kb) {
    __syncthreads();
    // Stage V^T: each thread reads 16B (8 hd-contig bf16), scatters to Vs.
    for (int j = 0; j < 4; ++j) {
      int c = tid + j * 256;           // 0..1023
      int kl = c >> 3, hc = (c & 7) * 8;
      uint4 vv = *(const uint4*)(v + base + (size_t)(kb * 128 + kl) * D_ + hc);
      const __bf16* e = (const __bf16*)&vv;
      for (int qd = 0; qd < 8; ++qd) Vs[hc + qd][kl] = e[qd];
    }
    __syncthreads();

    // S strip 16x128 = 8 tiles. B operands from global K rows, pipelined:
    // load tile t+1's two fragments before running tile t's WMMAs.
    Acc sacc[8];
    Frag bf[2][2];
    {
      const __bf16* kp = k + base + (size_t)(kb * 128 + lp) * D_;
      for (int kk = 0; kk < 2; ++kk) {
        bf[0][kk].u[0] = *(const uint4*)(kp + kk * 32 + half * 8);
        bf[0][kk].u[1] = *(const uint4*)(kp + kk * 32 + 16 + half * 8);
      }
    }
#pragma unroll
    for (int t = 0; t < 8; ++t) {
      const int curb = t & 1, nxtb = curb ^ 1;
      if (t < 7) {
        const __bf16* kp = k + base + (size_t)(kb * 128 + (t + 1) * 16 + lp) * D_;
        for (int kk = 0; kk < 2; ++kk) {
          bf[nxtb][kk].u[0] = *(const uint4*)(kp + kk * 32 + half * 8);
          bf[nxtb][kk].u[1] = *(const uint4*)(kp + kk * 32 + 16 + half * 8);
        }
      }
      v8f z = wmma_bf16(qf[0].v, bf[curb][0].v, v8f{0.f,0.f,0.f,0.f,0.f,0.f,0.f,0.f});
      z = wmma_bf16(qf[1].v, bf[curb][1].v, z);
      sacc[t].v = z;
    }

    // Online softmax. Row of C tile = i + 8*half; col = t*16 + lp.
    const bool diag = (kb == qb);
    float smax[8];
    for (int i = 0; i < 8; ++i) smax[i] = NEG;
    for (int t = 0; t < 8; ++t) {
      const int key = kb * 128 + t * 16 + lp;
      for (int i = 0; i < 8; ++i) {
        const int qr = qb * 128 + wave * 16 + i + 8 * half;
        float s = sacc[t].f[i] * scale;
        if (diag && key > qr) s = NEG;
        sacc[t].f[i] = s;
        smax[i] = fmaxf(smax[i], s);
      }
    }
    for (int d = 1; d < 16; d <<= 1)
      for (int i = 0; i < 8; ++i)
        smax[i] = fmaxf(smax[i], __shfl_xor(smax[i], d, 32));
    float alpha[8], ssum[8];
    for (int i = 0; i < 8; ++i) {
      float nm = fmaxf(mrow[i], smax[i]);
      alpha[i] = expf(mrow[i] - nm);
      mrow[i]  = nm;
      ssum[i]  = 0.f;
    }
    for (int t = 0; t < 8; ++t)
      for (int i = 0; i < 8; ++i) {
        float p = expf(sacc[t].f[i] - mrow[i]);
        sacc[t].f[i] = p;
        ssum[i] += p;
      }
    for (int d = 1; d < 16; d <<= 1)
      for (int i = 0; i < 8; ++i)
        ssum[i] += __shfl_xor(ssum[i], d, 32);
    for (int i = 0; i < 8; ++i) lrow[i] = lrow[i] * alpha[i] + ssum[i];
    for (int nt = 0; nt < 4; ++nt)
      for (int i = 0; i < 8; ++i) oacc[nt].f[i] *= alpha[i];

    // P: C-layout regs -> LDS -> A-layout frags (wave-local; DS is in-order).
    for (int t = 0; t < 8; ++t)
      for (int i = 0; i < 8; ++i)
        Ps[wave][i + 8 * half][t * 16 + lp] = (__bf16)sacc[t].f[i];
    Frag pa[4];
    const __bf16* pr = &Ps[wave][lp][0];
    for (int kk = 0; kk < 4; ++kk) {
      pa[kk].u[0] = *(const uint4*)(pr + kk * 32 + half * 8);
      pa[kk].u[1] = *(const uint4*)(pr + kk * 32 + 16 + half * 8);
    }
    // O += P @ V : 4 hd-tiles x 4 K-steps of 32 keys.
    for (int nt = 0; nt < 4; ++nt) {
      const __bf16* vr = &Vs[nt * 16 + lp][0];
      v8f z = oacc[nt].v;
      for (int kk = 0; kk < 4; ++kk) {
        Frag bv;
        bv.u[0] = *(const uint4*)(vr + kk * 32 + half * 8);
        bv.u[1] = *(const uint4*)(vr + kk * 32 + 16 + half * 8);
        z = wmma_bf16(pa[kk].v, bv.v, z);
      }
      oacc[nt].v = z;
    }
  }

  for (int nt = 0; nt < 4; ++nt)
    for (int i = 0; i < 8; ++i) {
      const int qr = qb * 128 + wave * 16 + i + 8 * half;
      o[base + (size_t)qr * D_ + nt * 16 + lp] = (__bf16)(oacc[nt].f[i] / lrow[i]);
    }
}

// ---------------------------------------------------------------------------
// Host-side orchestration
// ---------------------------------------------------------------------------
extern "C" void kernel_launch(void* const* d_in, const int* in_sizes, int n_in,
                              void* d_out, int out_size, void* d_ws, size_t ws_size,
                              hipStream_t stream) {
  (void)in_sizes; (void)n_in; (void)out_size; (void)ws_size;
  const int*   idx   = (const int*)d_in[0];
  const float* tok   = (const float*)d_in[1];
  const float* pos   = (const float*)d_in[2];
  const float* ln1_g = (const float*)d_in[3];
  const float* ln1_b = (const float*)d_in[4];
  const float* Wq    = (const float*)d_in[5];
  const float* bq    = (const float*)d_in[6];
  const float* Wk    = (const float*)d_in[7];
  const float* bk    = (const float*)d_in[8];
  const float* Wv    = (const float*)d_in[9];
  const float* bv    = (const float*)d_in[10];
  const float* Wo    = (const float*)d_in[11];
  const float* bo    = (const float*)d_in[12];
  const float* ln2_g = (const float*)d_in[13];
  const float* ln2_b = (const float*)d_in[14];
  const float* W1    = (const float*)d_in[15];
  const float* b1    = (const float*)d_in[16];
  const float* W2    = (const float*)d_in[17];
  const float* b2    = (const float*)d_in[18];
  const float* lnf_g = (const float*)d_in[19];
  const float* lnf_b = (const float*)d_in[20];
  const float* Whead = (const float*)d_in[21];
  const float* bhead = (const float*)d_in[22];
  float* logits = (float*)d_out;

  char* ws = (char*)d_ws;
  const size_t MB = 1024 * 1024;
  __bf16* wt   = (__bf16*)(ws + 0);        // 64 MiB reused W^T slot
  float*  x    = (float*)(ws + 64 * MB);   // 8 MiB residual stream
  __bf16* hbuf = (__bf16*)(ws + 72 * MB);  // 4 MiB LN output
  __bf16* qbuf = (__bf16*)(ws + 76 * MB);
  __bf16* kbuf = (__bf16*)(ws + 80 * MB);
  __bf16* vbuf = (__bf16*)(ws + 84 * MB);
  __bf16* obuf = (__bf16*)(ws + 88 * MB);
  __bf16* ff   = (__bf16*)(ws + 92 * MB);  // 16 MiB MLP hidden

  __bf16* Wqt = wt;
  __bf16* Wkt = wt + 1 * (size_t)D_ * D_;
  __bf16* Wvt = wt + 2 * (size_t)D_ * D_;
  __bf16* Wot = wt + 3 * (size_t)D_ * D_;
  __bf16* W1t = wt + 4 * (size_t)D_ * D_;
  __bf16* W2t = W1t + (size_t)D_ * F_;

  const dim3 tb(32, 8);
  const dim3 gD(D_ / 128, M_ / 128);

  embed_kernel<<<M_, 256, 0, stream>>>(idx, tok, pos, x);

  for (int i = 0; i < L_; ++i) {
    const size_t wdd = (size_t)i * D_ * D_;
    const size_t wdf = (size_t)i * D_ * F_;
    transpose_cvt_kernel<<<dim3(D_ / 32, D_ / 32), tb, 0, stream>>>(Wq + wdd, Wqt, D_, D_);
    transpose_cvt_kernel<<<dim3(D_ / 32, D_ / 32), tb, 0, stream>>>(Wk + wdd, Wkt, D_, D_);
    transpose_cvt_kernel<<<dim3(D_ / 32, D_ / 32), tb, 0, stream>>>(Wv + wdd, Wvt, D_, D_);
    transpose_cvt_kernel<<<dim3(D_ / 32, D_ / 32), tb, 0, stream>>>(Wo + wdd, Wot, D_, D_);
    transpose_cvt_kernel<<<dim3(F_ / 32, D_ / 32), tb, 0, stream>>>(W1 + wdf, W1t, D_, F_);
    transpose_cvt_kernel<<<dim3(D_ / 32, F_ / 32), tb, 0, stream>>>(W2 + wdf, W2t, F_, D_);

    layernorm_kernel<<<M_, 256, 0, stream>>>(x, ln1_g + i * D_, ln1_b + i * D_, hbuf);
    gemm_kernel<false, false, false, true><<<gD, 256, 0, stream>>>(
        hbuf, Wqt, bq + i * D_, nullptr, nullptr, qbuf, M_, D_, D_);
    gemm_kernel<false, false, false, true><<<gD, 256, 0, stream>>>(
        hbuf, Wkt, bk + i * D_, nullptr, nullptr, kbuf, M_, D_, D_);
    gemm_kernel<false, false, false, true><<<gD, 256, 0, stream>>>(
        hbuf, Wvt, bv + i * D_, nullptr, nullptr, vbuf, M_, D_, D_);

    attention_kernel<<<dim3(T_ / 128, H_, B_), 256, 0, stream>>>(qbuf, kbuf, vbuf, obuf);

    gemm_kernel<true, false, true, false><<<gD, 256, 0, stream>>>(
        obuf, Wot, bo + i * D_, x, x, nullptr, M_, D_, D_);

    layernorm_kernel<<<M_, 256, 0, stream>>>(x, ln2_g + i * D_, ln2_b + i * D_, hbuf);
    gemm_kernel<false, true, false, true><<<dim3(F_ / 128, M_ / 128), 256, 0, stream>>>(
        hbuf, W1t, b1 + i * F_, nullptr, nullptr, ff, M_, F_, D_);
    gemm_kernel<true, false, true, false><<<gD, 256, 0, stream>>>(
        ff, W2t, b2 + i * D_, x, x, nullptr, M_, D_, F_);
  }

  layernorm_kernel<<<M_, 256, 0, stream>>>(x, lnf_g, lnf_b, hbuf);
  transpose_cvt_kernel<<<dim3(V_ / 32, D_ / 32), tb, 0, stream>>>(Whead, wt, D_, V_);
  gemm_kernel<false, false, true, false><<<dim3(V_ / 128, M_ / 128), 256, 0, stream>>>(
      hbuf, wt, bhead, nullptr, logits, nullptr, M_, V_, D_);
}
